// ProtoDINO_16801912062564
// MI455X (gfx1250) — matrix operand order repaired and checked by reading, not compiled, not run on hardware
//
#include <hip/hip_runtime.h>

#define DIM      768
#define B_IMGS   32
#define NPATCH   1024
#define NROWS    (B_IMGS*NPATCH)   // 32768
#define NCLS     201
#define KP       5
#define NPROT    (NCLS*KP)         // 1005
#define NPROT_PAD 1024
#define GAMMA_F  0.9f
#define SK_EPS_F 0.05f
#define SK_ITERS 3
#define PCA_ITERS 10

typedef __bf16 bf16;
typedef __attribute__((ext_vector_type(16))) __bf16 v16bf;
typedef __attribute__((ext_vector_type(8)))  __bf16 v8bf;
typedef __attribute__((ext_vector_type(8)))  float  v8f;

// ---------------- workspace layout ----------------
constexpr size_t AL(size_t x){ return (x + 255) & ~(size_t)255; }
constexpr size_t OFF_PTN  = 0;                                            // pt_norm f32 [NROWS,DIM]
constexpr size_t OFF_PTNB = OFF_PTN  + AL((size_t)NROWS*DIM*4);           // pt_norm bf16
constexpr size_t OFF_PRN  = OFF_PTNB + AL((size_t)NROWS*DIM*2);           // proto_norm f32 [NPROT,DIM]
constexpr size_t OFF_PRNB = OFF_PRN  + AL((size_t)NPROT*DIM*4);           // proto_norm bf16 padded [NPROT_PAD,DIM]
constexpr size_t OFF_MEAN = OFF_PRNB + AL((size_t)NPROT_PAD*DIM*2);       // column sums [DIM]
constexpr size_t OFF_XCT  = OFF_MEAN + AL((size_t)DIM*4);                 // centered transpose bf16 [DIM,NROWS]
constexpr size_t OFF_G    = OFF_XCT  + AL((size_t)DIM*NROWS*2);           // Gram f32 [DIM,DIM]
constexpr size_t OFF_V    = OFF_G    + AL((size_t)DIM*DIM*4);             // eigvec [DIM]
constexpr size_t OFF_U    = OFF_V    + AL((size_t)DIM*4);                 // u [NROWS]
constexpr size_t OFF_MM   = OFF_U    + AL((size_t)NROWS*4);               // ordered min/max keys [2]
constexpr size_t OFF_PL   = OFF_MM   + AL(8);                             // patch labels [NROWS]
constexpr size_t OFF_NCNT = OFF_PL   + AL((size_t)NROWS*4);               // per-class counts [NCLS]
constexpr size_t OFF_PRES = OFF_NCNT + AL((size_t)NCLS*4);                // present flags [NCLS]
constexpr size_t OFF_Q    = OFF_PRES + AL((size_t)NCLS*4);                // sinkhorn q [NROWS,KP]
constexpr size_t OFF_TOT  = OFF_Q    + AL((size_t)NROWS*KP*4);            // per-class total [NCLS]
constexpr size_t OFF_COLS = OFF_TOT  + AL((size_t)NCLS*4);                // colsum [NPROT]
constexpr size_t OFF_ACC  = OFF_COLS + AL((size_t)NPROT*4);               // max acc keys [B,NPROT_PAD]
constexpr size_t OFF_PNEW = OFF_ACC  + AL((size_t)B_IMGS*NPROT_PAD*4);    // P_new [NPROT,DIM]

// ---------------- helpers ----------------
__device__ __forceinline__ unsigned f2ord(float f){
  unsigned u = __float_as_uint(f);
  return (u & 0x80000000u) ? ~u : (u | 0x80000000u);
}
__device__ __forceinline__ float ord2f(unsigned u){
  return (u & 0x80000000u) ? __uint_as_float(u & 0x7FFFFFFFu) : __uint_as_float(~u);
}
__device__ __forceinline__ float waveAllSum(float s){
  #pragma unroll
  for(int off=16; off; off>>=1) s += __shfl_xor(s, off, 32);
  return s;
}
__device__ __forceinline__ v8f vzero8(){
  v8f z;
  #pragma unroll
  for(int i=0;i<8;i++) z[i]=0.0f;
  return z;
}
// A-matrix 16-bit layout: half-wave k-chunks [kb..kb+7] and [kb+16..kb+23]
__device__ __forceinline__ v16bf load_a16(const bf16* __restrict__ p){
  v8bf lo = *(const v8bf*)(p);
  v8bf hi = *(const v8bf*)(p + 16);
  v16bf r;
  #pragma unroll
  for(int i=0;i<8;i++){ r[i]=lo[i]; r[8+i]=hi[i]; }
  return r;
}
// contiguous 16 K values per half-wave (unused; B operands are loaded with load_a16)
__device__ __forceinline__ v16bf load_b16(const bf16* __restrict__ p){
  return *(const v16bf*)(p);
}

// ---------------- utility kernels ----------------
__global__ void k_fill_u32(unsigned* __restrict__ p, unsigned v, int n){
  int i = blockIdx.x*blockDim.x + threadIdx.x;
  if(i < n) p[i] = v;
}
__global__ void k_init_mm(unsigned* __restrict__ mm){
  mm[0] = 0xFF800000u;  // key(+inf): min accumulator
  mm[1] = 0x007FFFFFu;  // key(-inf): max accumulator
}

// column sums of tokens (for mean)
__global__ void k_colsum_tokens(const float* __restrict__ x, float* __restrict__ msum){
  int d  = blockIdx.x*blockDim.x + threadIdx.x;   // 0..767 (grid.x=3)
  int r0 = blockIdx.y * 512;
  float s = 0.f;
  for(int r=0;r<512;r++) s += x[(size_t)(r0+r)*DIM + d];
  atomicAdd(&msum[d], s);
}

// L2-normalize patch rows -> f32 + bf16
__global__ void k_rownorm_patches(const float* __restrict__ x, float* __restrict__ ptn,
                                  bf16* __restrict__ ptnb){
  int w    = blockIdx.x*(blockDim.x>>5) + (threadIdx.x>>5);
  int lane = threadIdx.x & 31;
  const float* row = x + (size_t)w*DIM;
  float s = 0.f;
  for(int i=lane;i<DIM;i+=32){ float v=row[i]; s += v*v; }
  s = waveAllSum(s);
  float sc = 1.0f / fmaxf(sqrtf(s), 1e-12f);
  for(int i=lane;i<DIM;i+=32){
    float v = row[i]*sc;
    ptn [(size_t)w*DIM+i] = v;
    ptnb[(size_t)w*DIM+i] = (bf16)v;
  }
}

// L2-normalize prototype rows -> f32 + zero-padded bf16
__global__ void k_rownorm_protos(const float* __restrict__ p, float* __restrict__ prn,
                                 bf16* __restrict__ prnb){
  int w    = blockIdx.x*(blockDim.x>>5) + (threadIdx.x>>5);  // 0..1023
  int lane = threadIdx.x & 31;
  if(w < NPROT){
    const float* row = p + (size_t)w*DIM;
    float s = 0.f;
    for(int i=lane;i<DIM;i+=32){ float v=row[i]; s += v*v; }
    s = waveAllSum(s);
    float sc = 1.0f / fmaxf(sqrtf(s), 1e-12f);
    for(int i=lane;i<DIM;i+=32){
      float v = row[i]*sc;
      prn [(size_t)w*DIM+i] = v;
      prnb[(size_t)w*DIM+i] = (bf16)v;
    }
  } else {
    for(int i=lane;i<DIM;i+=32) prnb[(size_t)w*DIM+i] = (bf16)0.0f;
  }
}

// centered transpose: xcT[d, n] = bf16(x[n,d] - mean[d])
__global__ void k_center_transpose(const float* __restrict__ x, const float* __restrict__ msum,
                                   bf16* __restrict__ xcT){
  __shared__ float tile[32][33];
  int d0 = blockIdx.x*32, n0 = blockIdx.y*32;
  int tx = threadIdx.x, ty = threadIdx.y;
  #pragma unroll
  for(int i=0;i<4;i++){
    int n = n0 + ty + i*8;
    tile[ty + i*8][tx] = x[(size_t)n*DIM + d0 + tx];
  }
  __syncthreads();
  const float invN = 1.0f / (float)NROWS;
  #pragma unroll
  for(int i=0;i<4;i++){
    int d = d0 + ty + i*8;
    float m = msum[d]*invN;
    xcT[(size_t)d*NROWS + n0 + tx] = (bf16)(tile[tx][ty + i*8] - m);
  }
}

// ---------------- WMMA GEMM: logits + fused per-image max epilogue ----------------
// wave computes 16(M) x 64(N); grid = 2048 mtiles * 16 ntile4 / 8 waves = 4096 blocks
__global__ void k_gemm_logits(const bf16* __restrict__ A, const bf16* __restrict__ Bm,
                              unsigned* __restrict__ accmax){
  int w    = blockIdx.x*8 + (threadIdx.x>>5);
  int lane = threadIdx.x & 31;
  int nt4  = w & 15;
  int mt   = w >> 4;
  int lrow = lane & 15;
  int ka   = (lane>>4)*8;
  int kbase= (lane>>4)*16;
  const bf16* Ab = A + (size_t)(mt*16 + lrow)*DIM;
  const bf16* Bb[4];
  #pragma unroll
  for(int j=0;j<4;j++) Bb[j] = Bm + (size_t)(nt4*64 + j*16 + lrow)*DIM;
  v8f acc[4] = { vzero8(), vzero8(), vzero8(), vzero8() };
  for(int kt=0; kt<DIM/32; ++kt){
    v16bf a = load_a16(Ab + kt*32 + ka);
    __builtin_prefetch(Ab + kt*32 + ka + 64, 0, 1);
    #pragma unroll
    for(int j=0;j<4;j++){
      v16bf b = load_a16(Bb[j] + kt*32 + ka);
      acc[j] = __builtin_amdgcn_wmma_f32_16x16x32_bf16(false, a, false, b,
                                                       (short)0, acc[j], false, false);
    }
  }
  int img = mt >> 6;   // 64 m-tiles per image
  #pragma unroll
  for(int j=0;j<4;j++){
    float mx = acc[j][0];
    #pragma unroll
    for(int r=1;r<8;r++) mx = fmaxf(mx, acc[j][r]);
    mx = fmaxf(mx, __shfl_xor(mx, 16, 32));
    int col = nt4*64 + j*16 + lrow;
    if(lane < 16 && col < NPROT)
      atomicMax(&accmax[img*NPROT_PAD + col], f2ord(mx));
  }
}

// ---------------- WMMA GEMM: Gram matrix G = Xc^T Xc ----------------
// wave computes 16x64 of 768x768; 48 mtiles * 12 ntile4 = 576 waves = 72 blocks
__global__ void k_gemm_gram(const bf16* __restrict__ xcT, float* __restrict__ G){
  int w    = blockIdx.x*8 + (threadIdx.x>>5);
  int lane = threadIdx.x & 31;
  int nt4  = w % 12;
  int mt   = w / 12;
  int lrow = lane & 15;
  int ka   = (lane>>4)*8;
  int kbase= (lane>>4)*16;
  const bf16* Ab = xcT + (size_t)(mt*16 + lrow)*NROWS;
  const bf16* Bb[4];
  #pragma unroll
  for(int j=0;j<4;j++) Bb[j] = xcT + (size_t)(nt4*64 + j*16 + lrow)*NROWS;
  v8f acc[4] = { vzero8(), vzero8(), vzero8(), vzero8() };
  for(int kt=0; kt<NROWS/32; ++kt){
    v16bf a = load_a16(Ab + kt*32 + ka);
    __builtin_prefetch(Ab + kt*32 + ka + 64, 0, 1);
    #pragma unroll
    for(int j=0;j<4;j++){
      v16bf b = load_a16(Bb[j] + kt*32 + ka);
      acc[j] = __builtin_amdgcn_wmma_f32_16x16x32_bf16(false, a, false, b,
                                                       (short)0, acc[j], false, false);
    }
  }
  int mBase = mt*16 + 8*(lane>>4);
  #pragma unroll
  for(int j=0;j<4;j++){
    int col = nt4*64 + j*16 + lrow;
    #pragma unroll
    for(int r=0;r<8;r++)
      G[(size_t)(mBase + r)*DIM + col] = acc[j][r];
  }
}

// ---------------- power iteration (single workgroup) ----------------
__global__ void k_power(const float* __restrict__ G, float* __restrict__ vout){
  __shared__ float v[DIM];
  __shared__ float red[1024];
  int t = threadIdx.x;                 // 768 threads
  unsigned h = (unsigned)t*2654435761u + 12345u;
  h ^= h>>16; h *= 2246822519u; h ^= h>>13;
  v[t] = (float)(h & 0xFFFFFFu) * (1.0f/8388608.0f) - 1.0f;
  if(t < 256) red[768+t] = 0.0f;
  __syncthreads();
  red[t] = v[t]*v[t]; __syncthreads();
  for(int s=512;s>0;s>>=1){ if(t<s) red[t]+=red[t+s]; __syncthreads(); }
  float nrm = fmaxf(sqrtf(red[0]), 1e-12f);
  v[t] = v[t]/nrm;
  __syncthreads();
  for(int it=0; it<PCA_ITERS; ++it){
    const float* row = G + (size_t)t*DIM;
    float y = 0.f;
    for(int d=0; d<DIM; ++d) y += row[d]*v[d];
    __syncthreads();
    red[t] = y*y; __syncthreads();
    for(int s=512;s>0;s>>=1){ if(t<s) red[t]+=red[t+s]; __syncthreads(); }
    float n2 = fmaxf(sqrtf(red[0]), 1e-12f);
    v[t] = y/n2;
    __syncthreads();
  }
  vout[t] = v[t];
}

// u = Xc @ v, plus ordered-key atomic min/max
__global__ void k_compute_u(const float* __restrict__ x, const float* __restrict__ msum,
                            const float* __restrict__ v, float* __restrict__ u,
                            unsigned* __restrict__ mm){
  int w    = blockIdx.x*(blockDim.x>>5) + (threadIdx.x>>5);
  int lane = threadIdx.x & 31;
  const float invN = 1.0f/(float)NROWS;
  const float* row = x + (size_t)w*DIM;
  float s = 0.f;
  for(int d=lane; d<DIM; d+=32) s += (row[d] - msum[d]*invN) * v[d];
  s = waveAllSum(s);
  if(lane == 0){
    u[w] = s;
    unsigned k = f2ord(s);
    atomicMin(&mm[0], k);
    atomicMax(&mm[1], k);
  }
}

// patch labels + per-class counts
__global__ void k_labels(const float* __restrict__ u, const unsigned* __restrict__ mm,
                         const int* __restrict__ labels, int* __restrict__ pl,
                         int* __restrict__ ncnt){
  int n = blockIdx.x*blockDim.x + threadIdx.x;
  if(n >= NROWS) return;
  float umin = ord2f(mm[0]), umax = ord2f(mm[1]);
  // note: reference scales u but ||u||=1 after pca; sign of eigvec flips fg set --
  // unavoidable without matching jax PRNG; formula kept identical otherwise.
  float us = (u[n] - umin) / (umax - umin);
  int b = n >> 10;
  int lv = labels[b]; lv = lv < 0 ? 0 : (lv > NCLS-2 ? NCLS-2 : lv);
  int lbl = (us < 0.5f) ? lv : (NCLS-1);
  pl[n] = lbl;
  atomicAdd(&ncnt[lbl], 1);
}

__global__ void k_present(const int* __restrict__ labels, int* __restrict__ present){
  int t = threadIdx.x;
  if(t < B_IMGS){
    int lv = labels[t]; lv = lv < 0 ? 0 : (lv > NCLS-2 ? NCLS-2 : lv);
    present[lv] = 1;
  }
}

// per-row selected-class logits -> q = exp(L/eps); accumulate per-class total
__global__ void k_sel(const float* __restrict__ ptn, const float* __restrict__ prn,
                      const int* __restrict__ pl, float* __restrict__ q,
                      float* __restrict__ totsum){
  int w    = blockIdx.x*(blockDim.x>>5) + (threadIdx.x>>5);
  int lane = threadIdx.x & 31;
  int c = pl[w];
  const float* xr = ptn + (size_t)w*DIM;
  const float* p0 = prn + (size_t)c*KP*DIM;
  float s[KP] = {0,0,0,0,0};
  for(int d=lane; d<DIM; d+=32){
    float xv = xr[d];
    #pragma unroll
    for(int k=0;k<KP;k++) s[k] += xv * p0[(size_t)k*DIM + d];
  }
  #pragma unroll
  for(int k=0;k<KP;k++) s[k] = waveAllSum(s[k]);
  if(lane == 0){
    float rs = 0.f;
    #pragma unroll
    for(int k=0;k<KP;k++){
      float e = expf(s[k] * (1.0f/SK_EPS_F));
      q[(size_t)w*KP + k] = e;
      rs += e;
    }
    atomicAdd(&totsum[c], rs);
  }
}

__global__ void k_sk_divtot(float* __restrict__ q, const int* __restrict__ pl,
                            const float* __restrict__ totsum){
  int n = blockIdx.x*blockDim.x + threadIdx.x;
  if(n >= NROWS) return;
  float t = fmaxf(totsum[pl[n]], 1e-9f);
  #pragma unroll
  for(int k=0;k<KP;k++) q[(size_t)n*KP+k] /= t;
}

__global__ void k_sk_colsum(const float* __restrict__ q, const int* __restrict__ pl,
                            float* __restrict__ colsum){
  int n = blockIdx.x*blockDim.x + threadIdx.x;
  if(n >= NROWS) return;
  int c = pl[n];
  #pragma unroll
  for(int k=0;k<KP;k++) atomicAdd(&colsum[c*KP+k], q[(size_t)n*KP+k]);
}

__global__ void k_sk_norm(float* __restrict__ q, const int* __restrict__ pl,
                          const float* __restrict__ colsum, const int* __restrict__ ncnt){
  int n = blockIdx.x*blockDim.x + threadIdx.x;
  if(n >= NROWS) return;
  int c = pl[n];
  float tmp[KP]; float rs = 0.f;
  #pragma unroll
  for(int k=0;k<KP;k++){
    float qk = q[(size_t)n*KP+k] / (fmaxf(colsum[c*KP+k], 1e-9f) * (float)KP);
    tmp[k] = qk; rs += qk;
  }
  float ncf = fmaxf((float)ncnt[c], 1.0f);
  float inv = 1.0f / (fmaxf(rs, 1e-9f) * ncf);
  #pragma unroll
  for(int k=0;k<KP;k++) q[(size_t)n*KP+k] = tmp[k]*inv;
}

// P_new scatter: P_new[c,k,:] += (q*n_c) * pt_norm[n,:], one block per row
__global__ void k_scatter(const float* __restrict__ q, const int* __restrict__ pl,
                          const int* __restrict__ ncnt, const int* __restrict__ present,
                          const float* __restrict__ ptn, float* __restrict__ pnew){
  int n = blockIdx.x;
  int c = pl[n];
  if(!present[c]) return;           // uniform per block
  __shared__ float coef[KP];
  if(threadIdx.x < KP){
    float ncf = fmaxf((float)ncnt[c], 1.0f);
    coef[threadIdx.x] = q[(size_t)n*KP + threadIdx.x] * ncf;
  }
  __syncthreads();
  for(int d=threadIdx.x; d<DIM; d+=blockDim.x){
    float xv = ptn[(size_t)n*DIM + d];
    #pragma unroll
    for(int k=0;k<KP;k++)
      atomicAdd(&pnew[((size_t)c*KP+k)*DIM + d], coef[k]*xv);
  }
}

__global__ void k_final_protos(const float* __restrict__ proto, const float* __restrict__ pnew,
                               const int* __restrict__ present, float* __restrict__ out){
  int i = blockIdx.x*blockDim.x + threadIdx.x;
  if(i >= NPROT*DIM) return;
  int c = i / (KP*DIM);
  float p = proto[i];
  out[i] = present[c] ? (GAMMA_F*p + (1.0f-GAMMA_F)*pnew[i]) : p;
}

__global__ void k_final_logits(const unsigned* __restrict__ accmax,
                               float* __restrict__ out_cls, float* __restrict__ out_ipl){
  int t = blockIdx.x*blockDim.x + threadIdx.x;
  if(t >= B_IMGS*NCLS) return;
  int b = t / NCLS, c = t % NCLS;
  float sum = 0.f;
  #pragma unroll
  for(int k=0;k<KP;k++){
    float f = ord2f(accmax[b*NPROT_PAD + c*KP + k]);
    out_ipl[(size_t)t*KP + k] = f;
    sum += f;
  }
  out_cls[t] = sum;
}

// ---------------- launch ----------------
extern "C" void kernel_launch(void* const* d_in, const int* in_sizes, int n_in,
                              void* d_out, int out_size, void* d_ws, size_t ws_size,
                              hipStream_t stream){
  const float* tokens = (const float*)d_in[0];
  const float* protos = (const float*)d_in[1];
  const int*   labels = (const int*)d_in[2];
  float* out = (float*)d_out;
  char*  ws  = (char*)d_ws;

  float*    ptn     = (float*)   (ws + OFF_PTN);
  bf16*     ptnb    = (bf16*)    (ws + OFF_PTNB);
  float*    prn     = (float*)   (ws + OFF_PRN);
  bf16*     prnb    = (bf16*)    (ws + OFF_PRNB);
  float*    msum    = (float*)   (ws + OFF_MEAN);
  bf16*     xcT     = (bf16*)    (ws + OFF_XCT);
  float*    G       = (float*)   (ws + OFF_G);
  float*    vvec    = (float*)   (ws + OFF_V);
  float*    u       = (float*)   (ws + OFF_U);
  unsigned* mm      = (unsigned*)(ws + OFF_MM);
  int*      pl      = (int*)     (ws + OFF_PL);
  int*      ncnt    = (int*)     (ws + OFF_NCNT);
  int*      present = (int*)     (ws + OFF_PRES);
  float*    q       = (float*)   (ws + OFF_Q);
  float*    totsum  = (float*)   (ws + OFF_TOT);
  float*    colsum  = (float*)   (ws + OFF_COLS);
  unsigned* accmax  = (unsigned*)(ws + OFF_ACC);
  float*    pnew    = (float*)   (ws + OFF_PNEW);

  float* out_cls = out;                         // [32,201]
  float* out_ipl = out + B_IMGS*NCLS;           // [32,201,5]
  float* out_pro = out + B_IMGS*NCLS*(1+KP);    // [201,5,768]

  auto nb = [](int n){ return (n + 255)/256; };

  // init
  k_fill_u32<<<nb(DIM),256,0,stream>>>((unsigned*)msum, 0u, DIM);
  k_fill_u32<<<1,256,0,stream>>>((unsigned*)ncnt, 0u, NCLS);
  k_fill_u32<<<1,256,0,stream>>>((unsigned*)present, 0u, NCLS);
  k_fill_u32<<<1,256,0,stream>>>((unsigned*)totsum, 0u, NCLS);
  k_fill_u32<<<nb(NPROT*DIM),256,0,stream>>>((unsigned*)pnew, 0u, NPROT*DIM);
  k_fill_u32<<<nb(B_IMGS*NPROT_PAD),256,0,stream>>>(accmax, 0x007FFFFFu, B_IMGS*NPROT_PAD);
  k_init_mm<<<1,1,0,stream>>>(mm);

  // preprocessing
  k_colsum_tokens<<<dim3(3,64),256,0,stream>>>(tokens, msum);
  k_rownorm_patches<<<NROWS/8,256,0,stream>>>(tokens, ptn, ptnb);
  k_rownorm_protos<<<NPROT_PAD/8,256,0,stream>>>(protos, prn, prnb);
  k_center_transpose<<<dim3(DIM/32, NROWS/32),dim3(32,8),0,stream>>>(tokens, msum, xcT);

  // WMMA GEMMs
  k_gemm_logits<<<4096,256,0,stream>>>(ptnb, prnb, accmax);
  k_gemm_gram<<<72,256,0,stream>>>(xcT, G);

  // PCA
  k_power<<<1,DIM,0,stream>>>(G, vvec);
  k_compute_u<<<NROWS/8,256,0,stream>>>(tokens, msum, vvec, u, mm);
  k_labels<<<nb(NROWS),256,0,stream>>>(u, mm, labels, pl, ncnt);
  k_present<<<1,32,0,stream>>>(labels, present);

  // sinkhorn on sparse per-row assignments
  k_sel<<<NROWS/8,256,0,stream>>>(ptn, prn, pl, q, totsum);
  k_sk_divtot<<<nb(NROWS),256,0,stream>>>(q, pl, totsum);
  for(int it=0; it<SK_ITERS; ++it){
    k_fill_u32<<<nb(NPROT),256,0,stream>>>((unsigned*)colsum, 0u, NPROT);
    k_sk_colsum<<<nb(NROWS),256,0,stream>>>(q, pl, colsum);
    k_sk_norm<<<nb(NROWS),256,0,stream>>>(q, pl, colsum, ncnt);
  }

  // prototype update + outputs
  k_scatter<<<NROWS,256,0,stream>>>(q, pl, ncnt, present, ptn, pnew);
  k_final_protos<<<nb(NPROT*DIM),256,0,stream>>>(protos, pnew, present, out_pro);
  k_final_logits<<<nb(B_IMGS*NCLS),256,0,stream>>>(accmax, out_cls, out_ipl);
}
